// DeepVanillaRNN_1365799600434
// MI455X (gfx1250) — compile-verified, matrix operand order checked by reading, not run on hardware
//
#include <hip/hip_runtime.h>
#include <hip/hip_bf16.h>

// CDNA5 / gfx1250: wave32, WMMA 16x16x32 bf16 (f32 acc), async global->LDS copies.
typedef __attribute__((ext_vector_type(16))) __bf16 v16bf;
typedef __attribute__((ext_vector_type(8)))  float  v8f;

#define BK   32   // K-step per LDS stage (one wmma K)
#define LDSP 40   // padded LDS row stride in bf16 elems (80B, 16B-aligned, conflict-reduced)

__device__ __forceinline__ v16bf ld_frag16(const __bf16* p) {
  union { v16bf v; uint4 q[2]; } u;
  u.q[0] = *(const uint4*)(p);
  u.q[1] = *(const uint4*)(p + 8);
  return u.v;
}

// Async DMA of 16B/8B per lane from global to LDS; tracked by ASYNCcnt.
__device__ __forceinline__ void async_cp16(const __bf16* g, __bf16* l) {
  asm volatile("global_load_async_to_lds_b128 %0, %1, off"
               :: "v"((unsigned)(size_t)l),
                  "v"((unsigned long long)(size_t)g)
               : "memory");
}
__device__ __forceinline__ void async_cp8(const __bf16* g, __bf16* l) {
  asm volatile("global_load_async_to_lds_b64 %0, %1, off"
               :: "v"((unsigned)(size_t)l),
                  "v"((unsigned long long)(size_t)g)
               : "memory");
}

// C[M,N] = act( A[M,K](bf16) @ Bt[N,K]^T(bf16) + bias[N] (+ add[m*addStride + n]) )
// 256 threads = 8 waves (WGM x WGN). Per-wave tile = TI x TJ wmma 16x16 tiles.
// Double-buffered LDS; async global->LDS staging pipelined over the K loop.
template<int ACT, int BM_, int BN_, int WGM, int WGN>
__global__ __launch_bounds__(256) void gemm_bf16(
    const __bf16* __restrict__ A, const __bf16* __restrict__ Bt,
    const float* __restrict__ bias,
    const float* __restrict__ add, long addStride,
    __bf16* __restrict__ Cb, float* __restrict__ Cf,
    int M, int N, int K)
{
  constexpr int TI = BM_ / (WGM * 16);
  constexpr int TJ = BN_ / (WGN * 16);
  constexpr int CA = (BM_ * 4) / 256;        // 16B A-chunks per thread (2 or 1)
  constexpr int BCH = (BN_ * BK * 2) / 256;  // B-chunk bytes per thread (16 or 8)
  constexpr int ISSUES = CA + 1;             // async ops per wave per tile (uniform)

  __shared__ __bf16 As[2][BM_ * LDSP];
  __shared__ __bf16 Bs[2][BN_ * LDSP];

  const int tid  = threadIdx.x;
  const int m0   = blockIdx.y * BM_;
  const int n0   = blockIdx.x * BN_;
  const int w    = tid >> 5;
  const int lane = tid & 31;
  const int wm   = w / WGN;
  const int wn   = w % WGN;
  const int kh   = lane >> 4;   // which 16-wide K half this lane supplies
  const int lrow = lane & 15;   // row (A) / col (B) within the 16x16 tile

  v8f acc[TI][TJ];
  #pragma unroll
  for (int i = 0; i < TI; ++i)
    #pragma unroll
    for (int j = 0; j < TJ; ++j)
      #pragma unroll
      for (int r = 0; r < 8; ++r) acc[i][j][r] = 0.0f;

  // Stage tile at K-offset kk into LDS buffer `buf` (uniform ISSUES per wave).
  auto stage = [&](int kk, int buf) {
    #pragma unroll
    for (int r = 0; r < CA; ++r) {
      const int c = tid + r * 256;
      const int row = c >> 2, cg = c & 3;
      async_cp16(&A[(size_t)(m0 + row) * K + kk + cg * 8],
                 &As[buf][row * LDSP + cg * 8]);
    }
    if constexpr (BCH == 16) {
      const int row = tid >> 2, cg = tid & 3;
      async_cp16(&Bt[(size_t)(n0 + row) * K + kk + cg * 8],
                 &Bs[buf][row * LDSP + cg * 8]);
    } else {
      const int row = tid >> 3, cg = tid & 7;
      async_cp8(&Bt[(size_t)(n0 + row) * K + kk + cg * 4],
                &Bs[buf][row * LDSP + cg * 4]);
    }
  };

  const int nk = K / BK;
  stage(0, 0);

  for (int k = 0; k < nk; ++k) {
    const int buf = k & 1;
    if (k + 1 < nk) {
      stage((k + 1) * BK, buf ^ 1);
      // In-order completion: <= ISSUES outstanding leaves only tile k+1 in flight.
      asm volatile("s_wait_asynccnt %0" :: "n"(ISSUES) : "memory");
    } else {
      asm volatile("s_wait_asynccnt 0" ::: "memory");
    }
    __syncthreads();

    v16bf af[TI], bfr[TJ];
    #pragma unroll
    for (int ti = 0; ti < TI; ++ti)
      af[ti] = ld_frag16(&As[buf][(wm * (TI * 16) + ti * 16 + lrow) * LDSP + kh * 16]);
    #pragma unroll
    for (int tj = 0; tj < TJ; ++tj)
      bfr[tj] = ld_frag16(&Bs[buf][(wn * (TJ * 16) + tj * 16 + lrow) * LDSP + kh * 16]);

    #pragma unroll
    for (int ti = 0; ti < TI; ++ti)
      #pragma unroll
      for (int tj = 0; tj < TJ; ++tj)
        acc[ti][tj] = __builtin_amdgcn_wmma_f32_16x16x32_bf16(
            false, af[ti], false, bfr[tj], (short)0, acc[ti][tj], false, false);
    __syncthreads();   // all waves done reading buf before it is re-staged at k+2
  }

  // Epilogue: bias + optional addend + activation; bf16 and/or f32 stores.
  #pragma unroll
  for (int ti = 0; ti < TI; ++ti) {
    #pragma unroll
    for (int tj = 0; tj < TJ; ++tj) {
      const int tn  = n0 + wn * (TJ * 16) + tj * 16 + lrow;
      const int tm0 = m0 + wm * (TI * 16) + ti * 16 + kh * 8;
      const float bn = bias ? bias[tn] : 0.0f;
      #pragma unroll
      for (int r = 0; r < 8; ++r) {
        const int m = tm0 + r;
        float v = acc[ti][tj][r] + bn;
        if (add) v += add[(size_t)m * (size_t)addStride + tn];
        if (ACT == 1) v = v > 0.0f ? v : 0.0f;
        if (ACT == 2) v = tanhf(v);
        const size_t o = (size_t)m * N + tn;
        if (Cb) Cb[o] = (__bf16)v;
        if (Cf) Cf[o] = v;
      }
    }
  }
}

__global__ void k_cvt_f32_bf16(const float* __restrict__ x, __bf16* __restrict__ y, long n) {
  long i = (long)blockIdx.x * blockDim.x + threadIdx.x;
  if (i < n) y[i] = (__bf16)x[i];
}

// Wt[n*K + k] = (bf16) W[k*N + n]   (store B-operands column-major-of-original)
__global__ void k_transpose_to_bf16(const float* __restrict__ W, __bf16* __restrict__ Wt,
                                    int K, int N) {
  int i = blockIdx.x * blockDim.x + threadIdx.x;
  if (i < N * K) {
    int n = i / K, k = i - n * K;
    Wt[i] = (__bf16)W[(size_t)k * N + n];
  }
}

__global__ void k_zero_bf16(__bf16* __restrict__ y, int n) {
  int i = blockIdx.x * blockDim.x + threadIdx.x;
  if (i < n) y[i] = (__bf16)0.0f;
}

// Softmax over T per (b,h) and weighted sum of hs.  aw: [T, B*H] f32, hs: [T, B*H] bf16.
__global__ void k_attn_pool(const float* __restrict__ aw, const __bf16* __restrict__ hs,
                            float* __restrict__ out, int T_, int BH) {
  int i = blockIdx.x * blockDim.x + threadIdx.x;
  if (i >= BH) return;
  float mx = -3.402823466e38f;
  for (int t = 0; t < T_; ++t) mx = fmaxf(mx, aw[(size_t)t * BH + i]);
  float s = 0.0f, acc = 0.0f;
  for (int t = 0; t < T_; ++t) {
    float e = expf(aw[(size_t)t * BH + i] - mx);
    s += e;
    acc += e * (float)hs[(size_t)t * BH + i];
  }
  out[i] = acc / s;
}

static void launch_gemm(int act, bool small_, const __bf16* A, const __bf16* Bt,
                        const float* bias, const float* add, long addStride,
                        __bf16* Cb, float* Cf, int M, int N, int K, hipStream_t s)
{
  if (!small_) {
    dim3 g(N / 64, M / 128), b(256);
    switch (act) {
      case 0: gemm_bf16<0,128,64,4,2><<<g,b,0,s>>>(A,Bt,bias,add,addStride,Cb,Cf,M,N,K); break;
      case 1: gemm_bf16<1,128,64,4,2><<<g,b,0,s>>>(A,Bt,bias,add,addStride,Cb,Cf,M,N,K); break;
      default:gemm_bf16<2,128,64,4,2><<<g,b,0,s>>>(A,Bt,bias,add,addStride,Cb,Cf,M,N,K); break;
    }
  } else {
    dim3 g(N / 32, M / 64), b(256);
    switch (act) {
      case 0: gemm_bf16<0,64,32,4,2><<<g,b,0,s>>>(A,Bt,bias,add,addStride,Cb,Cf,M,N,K); break;
      case 1: gemm_bf16<1,64,32,4,2><<<g,b,0,s>>>(A,Bt,bias,add,addStride,Cb,Cf,M,N,K); break;
      default:gemm_bf16<2,64,32,4,2><<<g,b,0,s>>>(A,Bt,bias,add,addStride,Cb,Cf,M,N,K); break;
    }
  }
}

extern "C" void kernel_launch(void* const* d_in, const int* in_sizes, int n_in,
                              void* d_out, int out_size, void* d_ws, size_t ws_size,
                              hipStream_t stream)
{
  (void)in_sizes; (void)n_in; (void)out_size; (void)ws_size;
  constexpr int Bq = 256, Tq = 512, INq = 128, Hq = 256, W1q = 512, W2q = 512;
  constexpr long M1 = (long)Bq * Tq;   // 131072 rows

  const float* x     = (const float*)d_in[0];
  const float* h_b1  = (const float*)d_in[2];
  const float* h_b2  = (const float*)d_in[4];
  const float* h_b3  = (const float*)d_in[6];
  const float* i_b1  = (const float*)d_in[8];
  const float* i_b2  = (const float*)d_in[10];
  const float* i_b3  = (const float*)d_in[12];
  const float* att_b = (const float*)d_in[14];
  float* out = (float*)d_out;

  char* ws = (char*)d_ws;
  size_t off = 0;
  auto alloc = [&](size_t bytes) -> void* {
    void* p = ws + off;
    off = (off + bytes + 255) & ~(size_t)255;
    return p;
  };

  // Phase-1 scratch (dead after phase 1) — aliased by attention logits `aw`.
  const size_t scratch_base = off;
  __bf16* xbf = (__bf16*)alloc((size_t)M1 * INq * 2);
  __bf16* t1  = (__bf16*)alloc((size_t)M1 * W1q * 2);
  __bf16* t2  = (__bf16*)alloc((size_t)M1 * W2q * 2);
  float*  aw  = (float*)(ws + scratch_base);           // needs M1*Hq*4 <= scratch

  float*  xin  = (float*)alloc((size_t)M1 * Hq * 4);   // [B,T,H] f32 (rows = b*T+t)
  __bf16* hs   = (__bf16*)alloc((size_t)Tq * Bq * Hq * 2);
  __bf16* h0   = (__bf16*)alloc((size_t)Bq * Hq * 2);
  __bf16* a1   = (__bf16*)alloc((size_t)Bq * W1q * 2);
  __bf16* a2   = (__bf16*)alloc((size_t)Bq * W2q * 2);
  __bf16* iW1t = (__bf16*)alloc((size_t)W1q * INq * 2);
  __bf16* iW2t = (__bf16*)alloc((size_t)W2q * W1q * 2);
  __bf16* iW3t = (__bf16*)alloc((size_t)Hq  * W2q * 2);
  __bf16* hW1t = (__bf16*)alloc((size_t)W1q * Hq  * 2);
  __bf16* hW2t = (__bf16*)alloc((size_t)W2q * W1q * 2);
  __bf16* hW3t = (__bf16*)alloc((size_t)Hq  * W2q * 2);
  __bf16* aWt  = (__bf16*)alloc((size_t)Hq  * Hq  * 2);

  // ---- prep: bf16 conversion + weight transposes ----
  {
    long n = M1 * INq;
    k_cvt_f32_bf16<<<dim3((unsigned)((n + 255) / 256)), dim3(256), 0, stream>>>(x, xbf, n);
  }
  auto tr = [&](const float* W, __bf16* Wt, int K, int N) {
    int n = K * N;
    k_transpose_to_bf16<<<dim3((n + 255) / 256), dim3(256), 0, stream>>>(W, Wt, K, N);
  };
  tr((const float*)d_in[7],  iW1t, INq, W1q);
  tr((const float*)d_in[9],  iW2t, W1q, W2q);
  tr((const float*)d_in[11], iW3t, W2q, Hq);
  tr((const float*)d_in[1],  hW1t, Hq,  W1q);
  tr((const float*)d_in[3],  hW2t, W1q, W2q);
  tr((const float*)d_in[5],  hW3t, W2q, Hq);
  tr((const float*)d_in[13], aWt,  Hq,  Hq);
  k_zero_bf16<<<dim3((Bq * Hq + 255) / 256), dim3(256), 0, stream>>>(h0, Bq * Hq);

  // ---- phase 1: input MLP over all B*T rows (big parallel GEMMs) ----
  launch_gemm(1, false, xbf, iW1t, i_b1, nullptr, 0, t1, nullptr, (int)M1, W1q, INq, stream);
  launch_gemm(1, false, t1,  iW2t, i_b2, nullptr, 0, t2, nullptr, (int)M1, W2q, W1q, stream);
  launch_gemm(0, false, t2,  iW3t, i_b3, nullptr, 0, nullptr, xin, (int)M1, Hq, W2q, stream);

  // ---- phase 2: recurrence, 512 sequential small-GEMM steps ----
  for (int t = 0; t < Tq; ++t) {
    const __bf16* hin = (t == 0) ? h0 : (hs + (size_t)(t - 1) * Bq * Hq);
    launch_gemm(1, true, hin, hW1t, h_b1, nullptr, 0, a1, nullptr, Bq, W1q, Hq, stream);
    launch_gemm(1, true, a1,  hW2t, h_b2, nullptr, 0, a2, nullptr, Bq, W2q, W1q, stream);
    // h' = tanh(a2 @ hW3 + b3 + xin[:, t, :]); xin row for batch b is at (b*T + t).
    launch_gemm(2, true, a2,  hW3t, h_b3, xin + (size_t)t * Hq, (long)Tq * Hq,
                hs + (size_t)t * Bq * Hq, nullptr, Bq, Hq, W2q, stream);
  }

  // ---- phase 3: attention logits GEMM (+tanh) then softmax-pool over T ----
  launch_gemm(2, false, hs, aWt, att_b, nullptr, 0, nullptr, aw, (int)M1, Hq, Hq, stream);
  k_attn_pool<<<dim3((Bq * Hq + 255) / 256), dim3(256), 0, stream>>>(aw, hs, out, Tq, Bq * Hq);
}
